// RNNTPredictorJoiner_37658273251943
// MI455X (gfx1250) — compile-verified, matrix operand order checked by reading, not run on hardware
//
#include <hip/hip_runtime.h>
#include <hip/hip_bf16.h>

typedef __attribute__((ext_vector_type(8)))  _Float16 v8h;
typedef __attribute__((ext_vector_type(16))) _Float16 v16h;
typedef __attribute__((ext_vector_type(8)))  float    v8f;

#if __has_builtin(__builtin_amdgcn_tanhf)
#define DEV_TANHF(x) __builtin_amdgcn_tanhf(x)   // v_tanh_f32 (gfx1250 HW op)
#else
#define DEV_TANHF(x) tanhf(x)
#endif

// ---------------------------------------------------------------------------
// Weight pre-pack: f32 W[N x K] row-major  ->  f16 fragments laid out as
// P[ntile][ktile][lane][16], matching the V_WMMA_F32_16X16X32_F16 B-matrix
// register layout (lane = column n (mod 16); K striped: lanes 0-15 hold
// K {0..7,16..23}, lanes 16-31 hold K {8..15,24..31} of each 32-K slab).
// One wave packs one 16x32 tile; each lane emits one contiguous 32B fragment.
// ---------------------------------------------------------------------------
template <int K, int N>
__global__ __launch_bounds__(256) void pack_w(const float* __restrict__ W,
                                              _Float16* __restrict__ P) {
  constexpr int KT = K / 32;
  constexpr int NT = N / 16;
  const int wave = threadIdx.x >> 5;
  const int lane = threadIdx.x & 31;
  const int tile = blockIdx.x * 8 + wave;
  if (tile >= NT * KT) return;
  const int nt = tile / KT;
  const int kt = tile - nt * KT;
  const int n  = nt * 16 + (lane & 15);
  const int kb = kt * 32 + ((lane >> 4) << 3);
  v16h v;
#pragma unroll
  for (int i = 0; i < 16; ++i) {
    const int k = kb + (i & 7) + ((i >> 3) << 4);
    v[i] = (_Float16)W[(size_t)n * K + k];
  }
  *(v16h*)(P + ((size_t)tile * 32 + lane) * 16) = v;
}

// ---------------------------------------------------------------------------
// Register-blocked WMMA GEMM: out[MROWS x N] = act(A[MROWS x K]) * B^T + bias
//   MODE 0: A rows read directly from Aenc           (enc projection)
//   MODE 1: A rows gathered: emb_table[prefix[r]]    (pred projection)
//   MODE 2: A rows = tanh(enc_proj[b,t] + pred_proj[b,u])  (joint, fused)
// 8 waves; each wave owns N-tiles {wave, wave+8, ...} and iterates ALL
// MROWS/16 M-tiles per B fragment: one coalesced 32B B load feeds MT
// back-to-back WMMAs (register blocking on M => B/L2 traffic / MT).
// A tile staged in LDS with row stride K+8 halves: 324 dwords mod 64 banks
// = 4, so the 16 lanes' 4-dword ds_load_b128 chunks hit 64 distinct banks
// (conflict-free; unpadded K=512/640 strides are 0 mod 64 -> 16-way conflict).
// ---------------------------------------------------------------------------
template <int K, int N, int MODE, int MROWS>
__global__ __launch_bounds__(256) void gemm_wmma(
    const float* __restrict__ Aenc,      // MODE0: A base / MODE2: enc_proj
    const float* __restrict__ Apred,     // MODE1: emb_table / MODE2: pred_proj
    const int*   __restrict__ gatherIdx, // MODE1: prefix
    const _Float16* __restrict__ Bpack,
    const float* __restrict__ bias,
    float* __restrict__ out) {
  constexpr int KT = K / 32;
  constexpr int NT = N / 16;
  constexpr int MT = MROWS / 16;         // M-tiles register-blocked per wave
  constexpr int KP = K + 8;              // padded LDS row stride (bank-spread)
  __shared__ _Float16 As[MROWS * KP];    // 162 KB for MROWS=128,K=640

  const int tid  = threadIdx.x;
  const int row0 = blockIdx.x * MROWS;

  // ---- Phase 1: build f16 A tile (fused gather / broadcast-add / tanh) ----
  for (int i = tid; i < MROWS * K; i += 256) {
    const int row = i / K;
    const int col = i - row * K;
    const int r   = row0 + row;
    float v;
    if constexpr (MODE == 0) {
      v = Aenc[(size_t)r * K + col];
    } else if constexpr (MODE == 1) {
      v = Apred[(size_t)gatherIdx[r] * K + col];
    } else {
      const int b   = r / 20000;          // T*U = 20000
      const int rem = r - b * 20000;
      const int t   = rem / 100;          // U = 100
      const int u   = rem - t * 100;
      v = DEV_TANHF(Aenc[((size_t)(b * 200 + t)) * 640 + col] +
                    Apred[((size_t)(b * 100 + u)) * 640 + col]);
    }
    As[(size_t)row * KP + col] = (_Float16)v;
  }
  __syncthreads();

  // ---- Phase 2: WMMA with M register blocking ----
  const int lane = tid & 31;
  const int wave = tid >> 5;
  const int koff = (lane >> 4) << 3;      // 0 or 8 within a 32-K slab

  // Per-mtile LDS base for this lane's A row (lane&15 selects M within tile)
  const _Float16* aBase[MT];
#pragma unroll
  for (int mt = 0; mt < MT; ++mt)
    aBase[mt] = &As[(size_t)(mt * 16 + (lane & 15)) * KP];

  for (int nt = wave; nt < NT; nt += 8) {
    const int ncol = nt * 16;
    const float bj = bias[ncol + (lane & 15)];
    v8f acc[MT];
#pragma unroll
    for (int mt = 0; mt < MT; ++mt)
      acc[mt] = (v8f){bj, bj, bj, bj, bj, bj, bj, bj};

    const _Float16* bTile = Bpack + ((size_t)nt * KT * 32 + lane) * 16;
    __builtin_prefetch(bTile, 0, 1);      // global_prefetch_b8

    for (int kt = 0; kt < KT; ++kt) {
      // One B fragment (32B coalesced per lane) shared across MT M-tiles
      const v8h* bp = (const v8h*)(bTile + (size_t)kt * 32 * 16);
      const v8h b0 = bp[0];
      const v8h b1 = bp[1];
      const v16h bm = __builtin_shufflevector(b0, b1, 0, 1, 2, 3, 4, 5, 6, 7,
                                              8, 9, 10, 11, 12, 13, 14, 15);
      const int kb = kt * 32 + koff;
#pragma unroll
      for (int mt = 0; mt < MT; ++mt) {
        const v8h a0 = *(const v8h*)(aBase[mt] + kb);
        const v8h a1 = *(const v8h*)(aBase[mt] + kb + 16);
        const v16h a = __builtin_shufflevector(a0, a1, 0, 1, 2, 3, 4, 5, 6, 7,
                                               8, 9, 10, 11, 12, 13, 14, 15);
        acc[mt] = __builtin_amdgcn_wmma_f32_16x16x32_f16(
            /*neg_a=*/false, a, /*neg_b=*/false, bm,
            /*c_mod=*/(short)0, acc[mt], /*reuse_a=*/false, /*reuse_b=*/false);
      }
    }

    // C layout: VGPR vr, lanes 0-15 -> M=vr, lanes 16-31 -> M=vr+8; N=lane&15
#pragma unroll
    for (int mt = 0; mt < MT; ++mt) {
      float* op = out +
                  (size_t)(row0 + mt * 16 + ((lane >> 4) << 3)) * N +
                  ncol + (lane & 15);
#pragma unroll
      for (int vr = 0; vr < 8; ++vr) {
        if constexpr (MODE == 2) {
          __builtin_nontemporal_store(acc[mt][vr], op + (size_t)vr * N);
        } else {
          op[(size_t)vr * N] = acc[mt][vr];
        }
      }
    }
  }
}

// ---------------------------------------------------------------------------
// Host launch
// ---------------------------------------------------------------------------
extern "C" void kernel_launch(void* const* d_in, const int* in_sizes, int n_in,
                              void* d_out, int out_size, void* d_ws, size_t ws_size,
                              hipStream_t stream) {
  (void)in_sizes; (void)n_in; (void)out_size; (void)ws_size;
  constexpr int B = 8, T = 200, U = 100;
  constexpr int ENC = 512, EMB = 512, J = 640, V = 1024;

  const float* enc_out   = (const float*)d_in[0];  // [B,T,ENC]
  const int*   prefix    = (const int*)  d_in[1];  // [B,U]
  const float* emb_table = (const float*)d_in[2];  // [V,EMB]
  const float* W_enc     = (const float*)d_in[3];  // [J,ENC]
  const float* b_enc     = (const float*)d_in[4];  // [J]
  const float* W_pred    = (const float*)d_in[5];  // [J,EMB]
  const float* b_pred    = (const float*)d_in[6];  // [J]
  const float* W_join    = (const float*)d_in[7];  // [V,J]
  const float* b_join    = (const float*)d_in[8];  // [V]
  float* out = (float*)d_out;                      // [B,T,U,V]

  // Workspace layout (≈8.8 MB total)
  _Float16* pWenc  = (_Float16*)d_ws;              // J*ENC f16
  _Float16* pWpred = pWenc  + (size_t)J * ENC;     // J*EMB f16
  _Float16* pWjoin = pWpred + (size_t)J * EMB;     // V*J   f16
  float*    encP   = (float*)(pWjoin + (size_t)V * J);   // [B*T, J] f32
  float*    predP  = encP + (size_t)B * T * J;           // [B*U, J] f32

  // 1) pack weights to f16 WMMA-B fragment layout
  pack_w<ENC, J><<<(40 * 16 + 7) / 8, 256, 0, stream>>>(W_enc,  pWenc);
  pack_w<EMB, J><<<(40 * 16 + 7) / 8, 256, 0, stream>>>(W_pred, pWpred);
  pack_w<J, V>  <<<(64 * 20 + 7) / 8, 256, 0, stream>>>(W_join, pWjoin);

  // 2) enc projection: [B*T, ENC] x [ENC, J] -> encP   (1600 rows / 32)
  gemm_wmma<ENC, J, 0, 32><<<(B * T) / 32, 256, 0, stream>>>(
      enc_out, nullptr, nullptr, pWenc, b_enc, encP);

  // 3) pred projection with embedding gather: 800 rows / 32
  gemm_wmma<EMB, J, 1, 32><<<(B * U) / 32, 256, 0, stream>>>(
      nullptr, emb_table, prefix, pWpred, b_pred, predP);

  // 4) fused joint: tanh(encP + predP) x W_join^T + b_join -> out
  //    160,000 rows / 128 = 1250 workgroups; 128-row M-blocking cuts W_join
  //    L2 re-reads to ~1.6 GB, balancing the 655 MB HBM store floor.
  gemm_wmma<J, V, 2, 128><<<(B * T * U) / 128, 256, 0, stream>>>(
      encP, predP, nullptr, pWjoin, b_join, out);
}
